// GLBlock_61340722922003
// MI455X (gfx1250) — matrix operand order repaired and checked by reading, not compiled, not run on hardware
//
#include <hip/hip_runtime.h>
#include <hip/hip_bf16.h>
#include <cstddef>

// ---------------- WMMA types (CDNA5 gfx1250, wave32) ----------------
typedef __attribute__((ext_vector_type(16))) __bf16 v16bf;
typedef __attribute__((ext_vector_type(8)))  float  v8f;

union AFrag { v16bf v; uint4 q[2]; };
union BFrag { v16bf v; unsigned int w[8]; };

__device__ __forceinline__ v8f wmma_bf16(v16bf a, v16bf b, v8f c) {
  // D = A(16x32 bf16) * B(32x16 bf16) + C(16x16 f32)
  return __builtin_amdgcn_wmma_f32_16x16x32_bf16(false, a, false, b,
                                                 (short)0, c, false, false);
}

__device__ __forceinline__ unsigned short f2bf(float f) {
  unsigned u = __float_as_uint(f);
  u += 0x7FFFu + ((u >> 16) & 1u);   // round-to-nearest-even
  return (unsigned short)(u >> 16);
}

#define Hh 256
#define Ww 256
#define HW 65536
#define CPAD 224            // 196 -> 224 (7 K-slices of 32, 14 M-tiles of 16)
#define CP2  112            // channel pairs
#define HEADN (9*224*224)   // one 3x3 weight tensor, padded

// ---------------- K1: repack weights to padded bf16 A-layout ----------------
__global__ void prep_weights(const float* __restrict__ head_w,
                             const float* __restrict__ conv_w,
                             const float* __restrict__ last_w,
                             unsigned short* __restrict__ headA,
                             unsigned short* __restrict__ convA,
                             unsigned short* __restrict__ lastA) {
  int i = blockIdx.x * 256 + threadIdx.x;
  if (i < HEADN) {
    int tap = i / (224*224); int r = i % (224*224);
    int co = r / 224, ci = r % 224, kh = tap / 3, kw = tap % 3;
    float v = (co < 196 && ci < 196)
            ? head_w[(((size_t)co*196 + ci)*3 + kh)*3 + kw] : 0.f;
    headA[i] = f2bf(v); return;
  }
  i -= HEADN;
  if (i < 3*HEADN) {
    int l = i / HEADN; int r2 = i % HEADN;
    int tap = r2 / (224*224); int r = r2 % (224*224);
    int co = r / 224, ci = r % 224, kh = tap / 3, kw = tap % 3;
    float v = (co < 196 && ci < 196)
            ? conv_w[((((size_t)l*196 + co)*196 + ci)*3 + kh)*3 + kw] : 0.f;
    convA[i] = f2bf(v); return;
  }
  i -= 3*HEADN;
  if (i < 80*896) {
    int m = i / 896, k = i % 896, g = k / 224, cl = k % 224;
    float v = (m < 68 && cl < 196) ? last_w[(size_t)m*784 + g*196 + cl] : 0.f;
    lastA[i] = f2bf(v);
  }
}

// ---------------- K2: warp + concat + bf16 pair-pack ----------------
struct Gath { int i00, i01, i10, i11; float w00, w01, w10, w11; };

__global__ void warp_concat(const float* __restrict__ f0,
                            const float* __restrict__ f1,
                            const float* __restrict__ fd,
                            const float* __restrict__ fl,
                            unsigned int* __restrict__ xin) {
  int p = blockIdx.x * 256 + threadIdx.x;
  int y = p >> 8, x = p & 255;
  auto mk = [&](float px, float py) {
    Gath g;
    float x0 = floorf(px), y0 = floorf(py);
    float x1 = x0 + 1.f, y1 = y0 + 1.f;
    float wx1 = px - x0, wx0 = 1.f - wx1, wy1 = py - y0, wy0 = 1.f - wy1;
    float vx0 = (x0 >= 0.f && x0 <= 255.f) ? 1.f : 0.f;
    float vx1 = (x1 >= 0.f && x1 <= 255.f) ? 1.f : 0.f;
    float vy0 = (y0 >= 0.f && y0 <= 255.f) ? 1.f : 0.f;
    float vy1 = (y1 >= 0.f && y1 <= 255.f) ? 1.f : 0.f;
    int xi0 = (int)fminf(fmaxf(x0, 0.f), 255.f);
    int xi1 = (int)fminf(fmaxf(x1, 0.f), 255.f);
    int yi0 = (int)fminf(fmaxf(y0, 0.f), 255.f);
    int yi1 = (int)fminf(fmaxf(y1, 0.f), 255.f);
    g.i00 = yi0*256 + xi0; g.i01 = yi0*256 + xi1;
    g.i10 = yi1*256 + xi0; g.i11 = yi1*256 + xi1;
    g.w00 = wx0*wy0*vx0*vy0; g.w01 = wx1*wy0*vx1*vy0;
    g.w10 = wx0*wy1*vx0*vy1; g.w11 = wx1*wy1*vx1*vy1;
    return g;
  };
  Gath ga = mk((float)x + fl[p],         (float)y + fl[HW + p]);
  Gath gb = mk((float)x + fl[2*HW + p],  (float)y + fl[3*HW + p]);
  auto samp = [&](const float* img, const Gath& g) {
    return img[g.i00]*g.w00 + img[g.i01]*g.w01 + img[g.i10]*g.w10 + img[g.i11]*g.w11;
  };
  auto getc = [&](int c) -> float {
    if (c < 64)  return samp(f0 + (size_t)c*HW, ga);
    if (c < 128) return samp(f1 + (size_t)(c-64)*HW, gb);
    if (c < 192) return fd[(size_t)(c-128)*HW + p];
    if (c < 196) return fl[(size_t)(c-192)*HW + p];
    return 0.f;
  };
  for (int pr = 0; pr < CP2; ++pr) {
    float v0 = getc(2*pr), v1 = getc(2*pr + 1);
    xin[(size_t)pr*HW + p] = (unsigned)f2bf(v0) | ((unsigned)f2bf(v1) << 16);
  }
}

// ---------------- K3: 3x3 conv (implicit GEMM, WMMA bf16) ----------------
// in/out: pair-packed bf16 [112][HW]. Aw: [9][224 cout][224 cin] bf16.
// y = sm * prelu(conv3x3(in))   (== both head and body layer forms)
// 224 threads = 7 wave32; wave w owns M-tiles {w, w+7}; block covers 64 px.
// K split into two LDS phases (slices 0..3, 4..6) to keep LDS at 52 KB;
// 8 persistent accumulators per wave -> 8 WMMAs per K-slice.
__global__ __launch_bounds__(224) void conv3x3_wmma(
    const unsigned int* __restrict__ in,
    const unsigned short* __restrict__ Aw,
    const float* __restrict__ alpha,
    const float* __restrict__ sm,
    unsigned int* __restrict__ outp) {
  // 3 rows x (<=64 channel-pairs) x 66(+2 pad) pixels
  __shared__ unsigned int lds[3 * 64 * 68];
  const int tid = threadIdx.x;
  const int pixbase = blockIdx.x * 64;
  const int y = pixbase >> 8;
  const int xbase = pixbase & 255;

  const int wv = tid >> 5, lane = tid & 31, lo = lane & 15, hi = lane >> 4;
  const int mt0 = wv, mt1 = wv + 7;      // both always < 14

  v8f z = {0.f,0.f,0.f,0.f,0.f,0.f,0.f,0.f};
  v8f acc[2][4];
#pragma unroll
  for (int im = 0; im < 2; ++im)
#pragma unroll
    for (int nt = 0; nt < 4; ++nt) acc[im][nt] = z;

  const unsigned short* rowA0 = Aw + (size_t)(mt0*16 + lo)*CPAD + hi*16;
  const unsigned short* rowA1 = Aw + (size_t)(mt1*16 + lo)*CPAD + hi*16;

  for (int ph = 0; ph < 2; ++ph) {
    const int sLo = ph ? 4 : 0, sHi = ph ? 7 : 4;
    const int prBase = sLo * 16;           // first channel-pair in this phase
    const int nPr = (sHi - sLo) * 16;      // 64 or 48 pairs

    __syncthreads();   // previous phase's LDS reads done before refill
    for (int i = tid; i < 3 * nPr * 66; i += 224) {
      int xq = i % 66; int t = i / 66; int pl = t % nPr; int r = t / nPr;
      int gy = y + r - 1, gx = xbase + xq - 1;
      unsigned v = 0u;
      if ((unsigned)gy < 256u && (unsigned)gx < 256u)
        v = in[(size_t)(prBase + pl)*HW + (gy << 8) + gx];
      lds[(r*64 + pl)*68 + xq] = v;
    }
    __syncthreads();

    for (int kh = 0; kh < 3; ++kh) {
      for (int kw = 0; kw < 3; ++kw) {
        const int tap = kh*3 + kw;
        const size_t tapoff = (size_t)tap * CPAD * CPAD;
        // warm L2 for the next tap's A-fragment stream (global_prefetch_b8)
        if (tap + 1 < 9) {
          __builtin_prefetch(rowA0 + tapoff + (size_t)CPAD*CPAD, 0, 1);
          __builtin_prefetch(rowA1 + tapoff + (size_t)CPAD*CPAD, 0, 1);
        }
        for (int s = sLo; s < sHi; ++s) {
          const int kb = s * 32;
          AFrag A0, A1;
          const unsigned short* r0 = rowA0 + tapoff + kb;
          const unsigned short* r1 = rowA1 + tapoff + kb;
          A0.q[0] = *(const uint4*)(r0);
          A0.q[1] = *(const uint4*)(r0 + 8);
          A1.q[0] = *(const uint4*)(r1);
          A1.q[1] = *(const uint4*)(r1 + 8);

          const int plrow = kh*64 + (kb >> 1) - prBase + hi*8;
          BFrag B[4];
#pragma unroll
          for (int nt = 0; nt < 4; ++nt) {
            const int xq = nt*16 + lo + kw;
#pragma unroll
            for (int j = 0; j < 8; ++j)
              B[nt].w[j] = lds[(plrow + j)*68 + xq];
          }
#pragma unroll
          for (int nt = 0; nt < 4; ++nt) {
            acc[0][nt] = wmma_bf16(A0.v, B[nt].v, acc[0][nt]);
            acc[1][nt] = wmma_bf16(A1.v, B[nt].v, acc[1][nt]);
          }
        }
      }
    }
  }

  auto epi = [&](int mt, int nt, v8f a) {
    int pix = pixbase + nt*16 + lo;
    float smv = sm[pix];
#pragma unroll
    for (int j = 0; j < 8; j += 2) {
      int c0 = mt*16 + hi*8 + j;
      float al0 = alpha[c0     < 196 ? c0     : 195];
      float al1 = alpha[c0 + 1 < 196 ? c0 + 1 : 195];
      float v0 = a[j], v1 = a[j+1];
      v0 = (v0 >= 0.f ? v0 : al0*v0) * smv;
      v1 = (v1 >= 0.f ? v1 : al1*v1) * smv;
      outp[((size_t)(mt*8 + hi*4 + (j>>1)))*HW + pix] =
          (unsigned)f2bf(v0) | ((unsigned)f2bf(v1) << 16);
    }
  };
#pragma unroll
  for (int nt = 0; nt < 4; ++nt) {
    epi(mt0, nt, acc[0][nt]);
    epi(mt1, nt, acc[1][nt]);
  }
}

// ---------------- K4: 1x1 conv 896->68 (WMMA GEMM) + bias ----------------
__global__ __launch_bounds__(256) void conv1x1_wmma(
    const unsigned int* __restrict__ outs,   // [4*112][HW] pair-packed
    const unsigned short* __restrict__ Aw,   // [80][896]
    const float* __restrict__ bias,          // [68]
    float* __restrict__ xout) {              // [68][HW] f32
  const int tid = threadIdx.x, wv = tid >> 5, lane = tid & 31;
  const int lo = lane & 15, hi = lane >> 4;
  const int pixbase = blockIdx.x * 128 + wv * 16;

  v8f z = {0.f,0.f,0.f,0.f,0.f,0.f,0.f,0.f};
  v8f acc[5] = {z, z, z, z, z};

  for (int s = 0; s < 28; ++s) {
    const int kb = s * 32;
    BFrag B;
    const int prow = (kb >> 1) + hi*8;
#pragma unroll
    for (int j = 0; j < 8; ++j)
      B.w[j] = outs[(size_t)(prow + j)*HW + pixbase + lo];
#pragma unroll
    for (int mt = 0; mt < 5; ++mt) {
      AFrag A;
      const unsigned short* r = Aw + (size_t)(mt*16 + lo)*896 + kb + hi*16;
      A.q[0] = *(const uint4*)(r);
      A.q[1] = *(const uint4*)(r + 8);
      acc[mt] = wmma_bf16(A.v, B.v, acc[mt]);
    }
  }
#pragma unroll
  for (int mt = 0; mt < 5; ++mt) {
#pragma unroll
    for (int j = 0; j < 8; ++j) {
      int c = mt*16 + hi*8 + j;
      if (c < 68)
        xout[(size_t)c*HW + pixbase + lo] = acc[mt][j] + bias[c];
    }
  }
}

// ---------------- K5: tiny 2-channel 3x3 mask conv ----------------
__global__ void mask_conv(const float* __restrict__ xbuf,
                          const float* __restrict__ mw,
                          const float* __restrict__ mb,
                          float* __restrict__ smn) {
  int p = blockIdx.x * 256 + threadIdx.x;
  int y = p >> 8, x = p & 255;
  float a0 = mb[0], a1 = mb[1];
  for (int kh = 0; kh < 3; ++kh) {
    int gy = y + kh - 1; if ((unsigned)gy >= 256u) continue;
    for (int kw = 0; kw < 3; ++kw) {
      int gx = x + kw - 1; if ((unsigned)gx >= 256u) continue;
      const float* xp = xbuf + (size_t)4*HW + gy*256 + gx;
      for (int ic = 0; ic < 64; ++ic) {
        float xv = xp[(size_t)ic*HW];
        a0 += mw[((ic      )*3 + kh)*3 + kw] * xv;
        a1 += mw[((64 + ic )*3 + kh)*3 + kw] * xv;
      }
    }
  }
  smn[p] = a0; smn[HW + p] = a1;
}

// ---------------- K6: 2x bilinear upsample + mask compare ----------------
__global__ void upsample_out(const float* __restrict__ xbuf,
                             const float* __restrict__ smn,
                             const float* __restrict__ smask,
                             float* __restrict__ out) {
  const int QQ = 512 * 512;
  int q = blockIdx.x * 256 + threadIdx.x;
  int Y = q >> 9, X = q & 511;
  float sx = (float)X * 0.5f - 0.25f, sy = (float)Y * 0.5f - 0.25f;
  float fx0 = floorf(sx), fy0 = floorf(sy);
  float tx = sx - fx0, ty = sy - fy0;
  int x0 = (int)fx0, y0 = (int)fy0;
  int x0c = x0 < 0 ? 0 : (x0 > 255 ? 255 : x0);
  int x1 = x0 + 1; int x1c = x1 < 0 ? 0 : (x1 > 255 ? 255 : x1);
  int y0c = y0 < 0 ? 0 : (y0 > 255 ? 255 : y0);
  int y1 = y0 + 1; int y1c = y1 < 0 ? 0 : (y1 > 255 ? 255 : y1);
  float w00 = (1.f-tx)*(1.f-ty), w01 = tx*(1.f-ty);
  float w10 = (1.f-tx)*ty,       w11 = tx*ty;
  int i00 = y0c*256 + x0c, i01 = y0c*256 + x1c;
  int i10 = y1c*256 + x0c, i11 = y1c*256 + x1c;
  for (int c = 0; c < 68; ++c) {
    const float* ch = xbuf + (size_t)c*HW;
    out[(size_t)c*QQ + q] =
        ch[i00]*w00 + ch[i01]*w01 + ch[i10]*w10 + ch[i11]*w11;
  }
  float s0 = smn[i00]*w00 + smn[i01]*w01 + smn[i10]*w10 + smn[i11]*w11;
  const float* s1p = smn + HW;
  float s1 = s1p[i00]*w00 + s1p[i01]*w01 + s1p[i10]*w10 + s1p[i11]*w11;
  float m = (s0 > s1) ? 1.f : 0.f;
  out[(size_t)68*QQ + q] = m * smask[(Y >> 1)*256 + (X >> 1)];
}

// ---------------- launch ----------------
extern "C" void kernel_launch(void* const* d_in, const int* in_sizes, int n_in,
                              void* d_out, int out_size, void* d_ws, size_t ws_size,
                              hipStream_t stream) {
  (void)in_sizes; (void)n_in; (void)out_size; (void)ws_size;
  const float* feat_0  = (const float*)d_in[0];
  const float* feat_1  = (const float*)d_in[1];
  const float* feat_d  = (const float*)d_in[2];
  const float* flow    = (const float*)d_in[3];
  const float* smask   = (const float*)d_in[4];
  const float* head_w  = (const float*)d_in[5];
  const float* head_a  = (const float*)d_in[6];
  const float* conv_w  = (const float*)d_in[7];
  const float* conv_a  = (const float*)d_in[8];
  const float* last_w  = (const float*)d_in[9];
  const float* last_b  = (const float*)d_in[10];
  const float* mask_w  = (const float*)d_in[11];
  const float* mask_b  = (const float*)d_in[12];
  float* out = (float*)d_out;

  char* ws = (char*)d_ws;
  size_t off = 0;
  unsigned int*  xin   = (unsigned int*)(ws + off);  off += (size_t)CP2*HW*4;       // 29.4 MB
  unsigned int*  outs  = (unsigned int*)(ws + off);  off += (size_t)4*CP2*HW*4;     // 117.4 MB
  unsigned short* headA = (unsigned short*)(ws + off); off += (size_t)HEADN*2;
  unsigned short* convA = (unsigned short*)(ws + off); off += (size_t)3*HEADN*2;
  unsigned short* lastA = (unsigned short*)(ws + off); off += (size_t)80*896*2;
  float* xbuf = (float*)(ws + off);  off += (size_t)68*HW*4;                        // 17.8 MB
  float* smn  = (float*)(ws + off);  off += (size_t)2*HW*4;

  const int prepN = 4*HEADN + 80*896;
  prep_weights<<<(prepN + 255)/256, 256, 0, stream>>>(head_w, conv_w, last_w,
                                                      headA, convA, lastA);
  warp_concat<<<HW/256, 256, 0, stream>>>(feat_0, feat_1, feat_d, flow, xin);

  // head + 3 body layers: 1024 blocks x 224 threads (7 wave32 each, 64 px/block)
  conv3x3_wmma<<<1024, 224, 0, stream>>>(xin, headA, head_a, smask, outs);
  for (int l = 0; l < 3; ++l) {
    conv3x3_wmma<<<1024, 224, 0, stream>>>(outs + (size_t)l*CP2*HW,
                                           convA + (size_t)l*HEADN,
                                           conv_a + (size_t)l*196,
                                           smask,
                                           outs + (size_t)(l+1)*CP2*HW);
  }
  conv1x1_wmma<<<HW/128, 256, 0, stream>>>(outs, lastA, last_b, xbuf);
  mask_conv<<<HW/256, 256, 0, stream>>>(xbuf, mask_w, mask_b, smn);
  upsample_out<<<(512*512)/256, 256, 0, stream>>>(xbuf, smn, smask, out);
}
